// TransformerXLModel_32126355374567
// MI455X (gfx1250) — compile-verified, matrix operand order checked by reading, not run on hardware
//
#include <hip/hip_runtime.h>
#include <hip/hip_bf16.h>
#include <math.h>

// ---------------------------------------------------------------------------
// Transformer-XL forward for MI455X (gfx1250, wave32, WMMA).
// L=6 B=2 T=1024 C=1024 H=16 DH=64 V=32000 MEM=128
// Token count grows per layer: T_l = 1024 + 128*l, K_l = T_l + 128.
// All large GEMMs + attention contractions use v_wmma_f32_16x16x32_f16.
// GEMM LDS tiles are stored in fragment-native K-permuted layout so each
// lane's WMMA fragment is 32 contiguous bytes (2x ds_load_b128).
// A-tile staging uses GLOBAL_LOAD_ASYNC_TO_LDS (ASYNCcnt) when available.
// ---------------------------------------------------------------------------

typedef _Float16 v16h __attribute__((ext_vector_type(16)));
typedef _Float16 v8h  __attribute__((ext_vector_type(8)));
typedef float    v8f  __attribute__((ext_vector_type(8)));

#if defined(__HIP_DEVICE_COMPILE__) && \
    __has_builtin(__builtin_amdgcn_global_load_async_to_lds_b128) && \
    __has_builtin(__builtin_amdgcn_s_wait_asynccnt)
#define USE_ASYNC_COPY 1
typedef int v4i_b128 __attribute__((vector_size(16)));                 // b128 payload
typedef __attribute__((address_space(1))) v4i_b128 as1_v4i;            // global
typedef __attribute__((address_space(3))) v4i_b128 as3_v4i;            // LDS
#else
#define USE_ASYNC_COPY 0
#endif

#define CC   1024
#define HH   16
#define DHH  64
#define VV   32000
#define MEMN 128
#define LN_LAYERS 6
#define BB   2
#define TT0  1024

// K index interleave inside a 16-bit WMMA A/B fragment (ISA 7.12.2):
// VGPR pair i -> K offset {0,2,4,6,16,18,20,22}[i] + (lane<16 ? 0 : 8)
__device__ __forceinline__ int frag_k(int i, int kb) {
    return ((i < 4) ? (2 * i) : (16 + 2 * (i - 4))) + kb;
}

// Load a 16-element f16 fragment as two 16B LDS reads (ds_load_b128 x2).
__device__ __forceinline__ v16h ld_frag(const _Float16* p) {
    union { v16h v; v8h h[2]; } u;
    u.h[0] = *(const v8h*)(p);
    u.h[1] = *(const v8h*)(p + 8);
    return u.v;
}

// ---------------------------------------------------------------------------
// Embedding gather: x[b,t,:] = wte[idx[b,t],:]
// ---------------------------------------------------------------------------
__global__ void k_embed(const int* __restrict__ idx, const float* __restrict__ wte,
                        float* __restrict__ x, int total) {
    int i = blockIdx.x * blockDim.x + threadIdx.x;
    if (i >= total) return;
    int c = i & (CC - 1);
    int tok = i >> 10;
    x[i] = wte[(size_t)idx[tok] * CC + c];
}

// ---------------------------------------------------------------------------
// LayerNorm over C=1024. 256 threads/row. Writes f16 (GEMM input) at a
// remapped row ((row/T)*Kcat + rowoff + row%T) and optionally f32.
// ---------------------------------------------------------------------------
__global__ __launch_bounds__(256)
void k_ln(const float* __restrict__ in, const float* __restrict__ gw,
          const float* __restrict__ bw, _Float16* __restrict__ out16,
          float* __restrict__ out32, int T, int Kcat, int rowoff) {
    __shared__ float red[256];
    const int row = blockIdx.x;
    const int tid = threadIdx.x;
    const float* x = in + (size_t)row * CC;
    float v[4];
    float s = 0.f;
#pragma unroll
    for (int i = 0; i < 4; ++i) { v[i] = x[tid + i * 256]; s += v[i]; }
    red[tid] = s; __syncthreads();
    for (int o = 128; o > 0; o >>= 1) { if (tid < o) red[tid] += red[tid + o]; __syncthreads(); }
    float mean = red[0] * (1.f / CC);
    __syncthreads();
    s = 0.f;
#pragma unroll
    for (int i = 0; i < 4; ++i) { float d = v[i] - mean; s += d * d; }
    red[tid] = s; __syncthreads();
    for (int o = 128; o > 0; o >>= 1) { if (tid < o) red[tid] += red[tid + o]; __syncthreads(); }
    float rstd = rsqrtf(red[0] * (1.f / CC) + 1e-5f);
    const int bidx = row / T, tt = row % T;
    const size_t drow = ((size_t)bidx * Kcat + rowoff + tt) * CC;
#pragma unroll
    for (int i = 0; i < 4; ++i) {
        int c = tid + i * 256;
        float y = (v[i] - mean) * rstd * gw[c] + bw[c];
        out16[drow + c] = (_Float16)y;
        if (out32) out32[(size_t)row * CC + c] = y;
    }
}

// mems[l] (f32) -> f16 rows [0,128) of cat buffer, per batch.
__global__ void k_memcat(const float* __restrict__ mem_l, _Float16* __restrict__ cat16,
                         int Kcat, int total) {
    int i = blockIdx.x * blockDim.x + threadIdx.x;
    if (i >= total) return;
    int c = i & (CC - 1);
    int r = (i >> 10) & (MEMN - 1);
    int b = i >> 17;
    cat16[((size_t)b * Kcat + r) * CC + c] = (_Float16)mem_l[i];
}

// new_mem = last 128 rows of h (ln1 output, f32) per batch -> d_out region.
__global__ void k_newmem(const float* __restrict__ h, float* __restrict__ dst,
                         int T, int total) {
    int i = blockIdx.x * blockDim.x + threadIdx.x;
    if (i >= total) return;
    int c = i & (CC - 1);
    int r = (i >> 10) & (MEMN - 1);
    int b = i >> 17;
    dst[i] = h[((size_t)b * T + (T - MEMN) + r) * CC + c];
}

// ---------------------------------------------------------------------------
// Tiled WMMA GEMM: out = act(A_f16[MxK] @ B_f32->f16[KxN] + bias) (+resid)
// WG = 256 threads (8 waves), tile 128x128, K-step 32.
// Each wave: 32x64 C-slice = 2 A-frags x 4 B-frags = 8 WMMA / K-step.
//
// LDS layout (both tiles): K dimension permuted in 8-groups {0,2,1,3} so a
// lane's 16-element fragment (K = {0..7,16..23} or {8..15,24..31}) is 32
// contiguous bytes. Row stride 40 f16 = 80B (16B aligned, bank-conflict-free
// for 16-lane fragment reads: 20*fm mod 64 is a 16-cycle permutation).
// Group g -> f16 offset {0,16,8,24}[g] = ((g&1)<<4)|((g&2)<<2)  (branchless).
// The permutation moves whole 16B blocks, so the A tile (already f16) can be
// DMA'd directly into LDS with GLOBAL_LOAD_ASYNC_TO_LDS_B128.
// ---------------------------------------------------------------------------
__global__ __launch_bounds__(256)
void k_gemm(const _Float16* __restrict__ A, const float* __restrict__ Bw,
            const float* __restrict__ bias, const float* __restrict__ resid,
            float* __restrict__ outF32, _Float16* __restrict__ outF16,
            int M, int N, int K, int actGelu) {
    __shared__ _Float16 As[128 * 40];   // 128 rows x 32 K (permuted, +8 pad)
    __shared__ _Float16 Bs[128 * 40];   // 128 cols x 32 K (transposed+permuted)

    const int tid  = threadIdx.x;
    const int lane = tid & 31;
    const int wid  = tid >> 5;
    const int wrow = wid >> 1;              // 0..3 -> 32-row slice
    const int wcol = wid & 1;               // 0..1 -> 64-col slice
    const int m0   = blockIdx.y * 128;
    const int n0   = blockIdx.x * 128;
    const int fm   = lane & 15;
    const int fb   = (lane < 16) ? 0 : 16;  // fragment half select (f16 elems)

    // staging assignments (all loop-invariant)
    const int ar = tid >> 1;                // A row 0..127 (sync path)
    const int ah = tid & 1;                 // A k-half (16 f16 each)
    const int br = tid >> 3;                // B k-row 0..31
    const int bc = (tid & 7) * 16;          // B col start
    const int bg = br >> 3;                 // B k-group
    const int bp = (((bg & 1) << 4) | ((bg & 2) << 2)) | (br & 7); // {0,16,8,24}[bg]

#if USE_ASYNC_COPY
    // Async A staging: 512 16B blocks / tile, 2 per thread.
    // Block id -> (row, group); group g -> LDS byte offset {0,32,16,48}[g].
    const int arow0 = tid >> 2,        ag0 = tid & 3;
    const int arow1 = (tid + 256) >> 2, ag1 = (tid + 256) & 3;
    const int apb0 = ((ag0 & 1) << 5) | ((ag0 & 2) << 3);
    const int apb1 = ((ag1 & 1) << 5) | ((ag1 & 2) << 3);
    as3_v4i* adst0 = (as3_v4i*)(v4i_b128*)((char*)&As[arow0 * 40] + apb0);
    as3_v4i* adst1 = (as3_v4i*)(v4i_b128*)((char*)&As[arow1 * 40] + apb1);
#endif

    v8f acc[2][4] = {};

    for (int k0 = 0; k0 < K; k0 += 32) {
        __syncthreads();
#if USE_ASYNC_COPY
        {   // stage A via async DMA (no VGPR round trip)
            const _Float16* src0 = A + (size_t)(m0 + arow0) * K + k0 + ag0 * 8;
            const _Float16* src1 = A + (size_t)(m0 + arow1) * K + k0 + ag1 * 8;
            __builtin_amdgcn_global_load_async_to_lds_b128(
                (as1_v4i*)(const v4i_b128*)src0, adst0, 0, 0);
            __builtin_amdgcn_global_load_async_to_lds_b128(
                (as1_v4i*)(const v4i_b128*)src1, adst1, 0, 0);
            if (k0 + 32 < K)
                __builtin_prefetch((const void*)(src0 + 32), 0, 1);
        }
#else
        {   // stage A: 32B/thread, permuted K groups
            const _Float16* src = A + (size_t)(m0 + ar) * K + k0 + ah * 16;
            v8h lo = *(const v8h*)(src);        // k rel 0..7   of this half
            v8h hi = *(const v8h*)(src + 8);    // k rel 8..15  of this half
            *(v8h*)&As[ar * 40 + ah * 8]      = lo;   // groups 0/2 -> p 0/8
            *(v8h*)&As[ar * 40 + 16 + ah * 8] = hi;   // groups 1/3 -> p 16/24
            if (k0 + 32 < K) __builtin_prefetch((const void*)(src + 32), 0, 1);
        }
#endif
        {   // stage B: 64B/thread coalesced f32 read, f16 transpose-scatter
            const float* src = Bw + (size_t)(k0 + br) * N + n0 + bc;
#pragma unroll
            for (int i = 0; i < 16; i += 4) {
                float4 f = *(const float4*)(src + i);
                Bs[(bc + i + 0) * 40 + bp] = (_Float16)f.x;
                Bs[(bc + i + 1) * 40 + bp] = (_Float16)f.y;
                Bs[(bc + i + 2) * 40 + bp] = (_Float16)f.z;
                Bs[(bc + i + 3) * 40 + bp] = (_Float16)f.w;
            }
            if (k0 + 32 < K) __builtin_prefetch((const void*)(src + (size_t)32 * N), 0, 1);
        }
#if USE_ASYNC_COPY
        __builtin_amdgcn_s_wait_asynccnt(0);
#endif
        __syncthreads();

        v16h af0 = ld_frag(&As[(wrow * 32 + fm) * 40 + fb]);
        v16h af1 = ld_frag(&As[(wrow * 32 + 16 + fm) * 40 + fb]);
#pragma unroll
        for (int j = 0; j < 4; ++j) {
            v16h bf = ld_frag(&Bs[(wcol * 64 + j * 16 + fm) * 40 + fb]);
            acc[0][j] = __builtin_amdgcn_wmma_f32_16x16x32_f16(
                false, af0, false, bf, (short)0, acc[0][j], false, false);
            acc[1][j] = __builtin_amdgcn_wmma_f32_16x16x32_f16(
                false, af1, false, bf, (short)0, acc[1][j], false, false);
        }
    }

    // Epilogue (tiling is exact for all GEMMs in this model: no bounds checks)
#pragma unroll
    for (int rr = 0; rr < 2; ++rr) {
#pragma unroll
        for (int j = 0; j < 4; ++j) {
#pragma unroll
            for (int r = 0; r < 8; ++r) {
                int row = m0 + wrow * 32 + rr * 16 + r + ((lane < 16) ? 0 : 8);
                int col = n0 + wcol * 64 + j * 16 + fm;
                float v = acc[rr][j][r];
                if (bias)    v += bias[col];
                if (actGelu) v = 0.5f * v * (1.0f + erff(v * 0.70710678118654752f));
                if (resid)   v += resid[(size_t)row * N + col];
                if (outF32)  outF32[(size_t)row * N + col] = v;
                if (outF16)  outF16[(size_t)row * N + col] = (_Float16)v;
            }
        }
    }
}

// ---------------------------------------------------------------------------
// Attention: one wave per 16-query tile per (b,h). Full score row (16 x K)
// lives in dynamic LDS (<=115KB; CDNA5 has 320KB/WGP).
// AC = (q+rbias) @ k^T via WMMA; BD from rel_shift of s_i = (q+rbias).remb
// (analytically: i in {q, q+1}); softmax; then P @ V via WMMA.
// ---------------------------------------------------------------------------
__global__ void k_attn(const float* __restrict__ qkv, const float* __restrict__ remb,
                       const float* __restrict__ rbias, _Float16* __restrict__ out16,
                       int Kt) {
    extern __shared__ float smem[];
    float* scores = smem;                 // 16 * Kt
    float* srow   = smem + 16 * Kt;       // 17 (+pad)

    const int lane = threadIdx.x;
    const int q0 = blockIdx.x * 16;
    const int h  = blockIdx.y;
    const int b  = blockIdx.z;
    const int C3 = 3 * CC;
    const float* base = qkv + (size_t)b * Kt * C3;
    const int fm = lane & 15;
    const int kb = (lane < 16) ? 0 : 8;
    const float* rb = rbias + h * DHH;
    const float* re = remb  + h * DHH;

    // Q fragments (DH=64 -> two K=32 chunks), rbias fused.
    v16h qf[2];
#pragma unroll
    for (int c = 0; c < 2; ++c) {
        const float* qp = base + (size_t)(q0 + fm) * C3 + h * DHH + c * 32;
        const float* rp = rb + c * 32;
#pragma unroll
        for (int i = 0; i < 8; ++i) {
            int d = frag_k(i, kb);
            qf[c][2 * i]     = (_Float16)(qp[d] + rp[d]);
            qf[c][2 * i + 1] = (_Float16)(qp[d + 1] + rp[d + 1]);
        }
    }

    // s_i for global rows q0..q0+16 (rel-shift only ever needs i = q or q+1)
    if (lane < 17) {
        int qg = q0 + lane;
        float s = 0.f;
        if (qg < Kt) {
            const float* qp = base + (size_t)qg * C3 + h * DHH;
            for (int d = 0; d < DHH; ++d) s += (qp[d] + rb[d]) * re[d];
        }
        srow[lane] = s;
    }
    __syncthreads();

    // AC scores: 2 WMMA per 16-key tile
    for (int kt = 0; kt < Kt; kt += 16) {
        v8f acc = {};
#pragma unroll
        for (int c = 0; c < 2; ++c) {
            v16h kf;
            const float* kp = base + (size_t)(kt + fm) * C3 + CC + h * DHH + c * 32;
#pragma unroll
            for (int i = 0; i < 8; ++i) {
                int d = frag_k(i, kb);
                kf[2 * i]     = (_Float16)kp[d];
                kf[2 * i + 1] = (_Float16)kp[d + 1];
            }
            acc = __builtin_amdgcn_wmma_f32_16x16x32_f16(
                false, qf[c], false, kf, (short)0, acc, false, false);
        }
#pragma unroll
        for (int r = 0; r < 8; ++r) {
            int m = r + ((lane < 16) ? 0 : 8);
            scores[m * Kt + kt + fm] = acc[r];
        }
    }
    __syncthreads();

    // softmax((AC + BD) / sqrt(DH)) per row; BD[q][k] = (j==0 ? 0 : s_i),
    // t=(q+1)*Kt+k, i=t/(Kt+1), j=t%(Kt+1); i-q0 in [0,16].
    for (int m = 0; m < 16; ++m) {
        int qg = q0 + m;
        float mx = -3.4e38f;
        for (int c = lane; c < Kt; c += 32) {
            int t = (qg + 1) * Kt + c;
            int i = t / (Kt + 1);
            int j = t - i * (Kt + 1);
            float bd = (j == 0) ? 0.f : srow[i - q0];
            float v = (scores[m * Kt + c] + bd) * 0.125f;   // 1/sqrt(64)
            scores[m * Kt + c] = v;
            mx = fmaxf(mx, v);
        }
#pragma unroll
        for (int o = 16; o > 0; o >>= 1) mx = fmaxf(mx, __shfl_xor(mx, o));
        float sum = 0.f;
        for (int c = lane; c < Kt; c += 32) {
            float e = __expf(scores[m * Kt + c] - mx);
            scores[m * Kt + c] = e;
            sum += e;
        }
#pragma unroll
        for (int o = 16; o > 0; o >>= 1) sum += __shfl_xor(sum, o);
        float inv = 1.f / sum;
        for (int c = lane; c < Kt; c += 32) scores[m * Kt + c] *= inv;
    }
    __syncthreads();

    // P @ V: 4 WMMA per 32-key chunk (DH=64 -> 4 N-tiles)
    v8f acc2[4] = {};
    for (int kt = 0; kt < Kt; kt += 32) {
        v16h pf;
#pragma unroll
        for (int i = 0; i < 8; ++i) {
            int kk = frag_k(i, kb);
            pf[2 * i]     = (_Float16)scores[fm * Kt + kt + kk];
            pf[2 * i + 1] = (_Float16)scores[fm * Kt + kt + kk + 1];
        }
#pragma unroll
        for (int j = 0; j < 4; ++j) {
            v16h vf;
            const float* vp = base + 2 * CC + h * DHH + j * 16 + fm;
#pragma unroll
            for (int i = 0; i < 8; ++i) {
                int kk = frag_k(i, kb);
                vf[2 * i]     = (_Float16)vp[(size_t)(kt + kk) * C3];
                vf[2 * i + 1] = (_Float16)vp[(size_t)(kt + kk + 1) * C3];
            }
            acc2[j] = __builtin_amdgcn_wmma_f32_16x16x32_f16(
                false, pf, false, vf, (short)0, acc2[j], false, false);
        }
    }
#pragma unroll
    for (int j = 0; j < 4; ++j) {
#pragma unroll
        for (int r = 0; r < 8; ++r) {
            int row = q0 + r + ((lane < 16) ? 0 : 8);
            int col = h * DHH + j * 16 + fm;
            out16[((size_t)b * Kt + row) * CC + col] = (_Float16)acc2[j][r];
        }
    }
}

// ---------------------------------------------------------------------------
// Host orchestration
// ---------------------------------------------------------------------------
extern "C" void kernel_launch(void* const* d_in, const int* in_sizes, int n_in,
                              void* d_out, int out_size, void* d_ws, size_t ws_size,
                              hipStream_t stream) {
    (void)in_sizes; (void)n_in; (void)out_size; (void)ws_size;

    const int*   idx    = (const int*)  d_in[0];
    const float* mems   = (const float*)d_in[1];
    const float* wte    = (const float*)d_in[2];
    const float* ln1_g  = (const float*)d_in[3];
    const float* ln1_b  = (const float*)d_in[4];
    const float* attn_w = (const float*)d_in[5];
    const float* attn_b = (const float*)d_in[6];
    const float* proj_w = (const float*)d_in[7];
    const float* proj_b = (const float*)d_in[8];
    const float* r_emb  = (const float*)d_in[9];
    const float* r_bias = (const float*)d_in[10];
    const float* ln2_g  = (const float*)d_in[11];
    const float* ln2_b  = (const float*)d_in[12];
    const float* fc_w   = (const float*)d_in[13];
    const float* fc_b   = (const float*)d_in[14];
    const float* fc2_w  = (const float*)d_in[15];
    const float* fc2_b  = (const float*)d_in[16];
    const float* lnf_g  = (const float*)d_in[17];
    const float* lnf_b  = (const float*)d_in[18];
    const float* lm_w   = (const float*)d_in[19];
    float* out = (float*)d_out;

    const int Tfin = TT0 + LN_LAYERS * MEMN;          // 1792
    const int Mmax = BB * Tfin;                        // 3584
    const size_t LOGITS = (size_t)BB * Tfin * VV;      // 114,688,000 floats

    // workspace carve-up (~150 MB)
    char* ws = (char*)d_ws;
    size_t off = 0;
    auto alloc = [&](size_t bytes) { size_t r = off; off += (bytes + 255) & ~(size_t)255; return r; };
    float*    xA    = (float*)   (ws + alloc((size_t)Mmax * CC * 4));
    float*    xB    = (float*)   (ws + alloc((size_t)Mmax * CC * 4));
    float*    hbuf  = (float*)   (ws + alloc((size_t)Mmax * CC * 4));      // ln1 f32 (new_mem)
    _Float16* cat16 = (_Float16*)(ws + alloc((size_t)Mmax * CC * 2));
    float*    qkvb  = (float*)   (ws + alloc((size_t)Mmax * 3 * CC * 4));
    _Float16* att16 = (_Float16*)(ws + alloc((size_t)Mmax * CC * 2));
    float*    yb    = (float*)   (ws + alloc((size_t)Mmax * CC * 4));
    _Float16* h216  = (_Float16*)(ws + alloc((size_t)Mmax * CC * 2));
    _Float16* g16   = (_Float16*)(ws + alloc((size_t)Mmax * 4 * CC * 2));
    _Float16* xf16  = (_Float16*)(ws + alloc((size_t)Mmax * CC * 2));

    // attention needs >64KB dynamic LDS (CDNA5: 320KB/WGP)
    (void)hipFuncSetAttribute(reinterpret_cast<const void*>(&k_attn),
                              hipFuncAttributeMaxDynamicSharedMemorySize, 120 * 1024);

    // embedding
    {
        int tot = BB * TT0 * CC;
        k_embed<<<(tot + 255) / 256, 256, 0, stream>>>(idx, wte, xA, tot);
    }

    float* xcur = xA;
    float* xnxt = xB;
    int T = TT0;

    for (int l = 0; l < LN_LAYERS; ++l) {
        const int Kt = T + MEMN;
        const int M  = BB * Kt;

        // ln1: f16 rows into cat16 at per-batch offset MEMN, f32 into hbuf
        k_ln<<<BB * T, 256, 0, stream>>>(xcur, ln1_g + l * CC, ln1_b + l * CC,
                                         cat16, hbuf, T, Kt, MEMN);
        // mems -> cat16 rows [0,128); new_mem output (last 128 rows of ln1)
        {
            int tot = BB * MEMN * CC;
            k_memcat<<<(tot + 255) / 256, 256, 0, stream>>>(
                mems + (size_t)l * BB * MEMN * CC, cat16, Kt, tot);
            k_newmem<<<(tot + 255) / 256, 256, 0, stream>>>(
                hbuf, out + LOGITS + (size_t)l * BB * MEMN * CC, T, tot);
        }
        // qkv = cat @ attn_w + attn_b
        {
            dim3 g(3 * CC / 128, M / 128);
            k_gemm<<<g, 256, 0, stream>>>(cat16, attn_w + (size_t)l * CC * 3 * CC,
                                          attn_b + (size_t)l * 3 * CC, nullptr,
                                          qkvb, nullptr, M, 3 * CC, CC, 0);
        }
        // attention -> att16 (f16, direct GEMM input)
        {
            dim3 g(Kt / 16, HH, BB);
            size_t sh = (size_t)(16 * Kt + 32) * sizeof(float);
            k_attn<<<g, 32, sh, stream>>>(qkvb, r_emb + (size_t)l * HH * DHH,
                                          r_bias + (size_t)l * HH * DHH, att16, Kt);
        }
        // y = att @ proj_w + proj_b
        {
            dim3 g(CC / 128, M / 128);
            k_gemm<<<g, 256, 0, stream>>>(att16, proj_w + (size_t)l * CC * CC,
                                          proj_b + (size_t)l * CC, nullptr,
                                          yb, nullptr, M, CC, CC, 0);
        }
        // ln2(y) -> h216
        k_ln<<<M, 256, 0, stream>>>(yb, ln2_g + l * CC, ln2_b + l * CC,
                                    h216, nullptr, M, M, 0);
        // g = gelu(h2 @ fc_w + fc_b)  (f16 out)
        {
            dim3 g(4 * CC / 128, M / 128);
            k_gemm<<<g, 256, 0, stream>>>(h216, fc_w + (size_t)l * CC * 4 * CC,
                                          fc_b + (size_t)l * 4 * CC, nullptr,
                                          nullptr, g16, M, 4 * CC, CC, 1);
        }
        // x' = y + (g @ fc2_w + fc2_b)
        {
            dim3 g(CC / 128, M / 128);
            k_gemm<<<g, 256, 0, stream>>>(g16, fc2_w + (size_t)l * 4 * CC * CC,
                                          fc2_b + (size_t)l * CC, yb,
                                          xnxt, nullptr, M, CC, 4 * CC, 0);
        }
        T = Kt;
        float* tmp = xcur; xcur = xnxt; xnxt = tmp;
    }

    // final LN + LM head
    {
        const int M = BB * T;   // 3584
        k_ln<<<M, 256, 0, stream>>>(xcur, lnf_g, lnf_b, xf16, nullptr, M, M, 0);
        dim3 g(VV / 128, M / 128);
        k_gemm<<<g, 256, 0, stream>>>(xf16, lm_w, nullptr, nullptr,
                                      out, nullptr, M, VV, CC, 0);
    }
}
